// SANet_6408091205717
// MI455X (gfx1250) — matrix operand order repaired and checked by reading, not compiled, not run on hardware
//
#include <hip/hip_runtime.h>

// ---------------------------------------------------------------------------
// Types for CDNA5 WMMA (wave32): v_wmma_f32_16x16x32_bf16
// ---------------------------------------------------------------------------
typedef __attribute__((ext_vector_type(16))) __bf16 v16bf;
typedef __attribute__((ext_vector_type(8)))  __bf16 v8bf;
typedef __attribute__((ext_vector_type(8)))  float  v8f;
typedef __attribute__((ext_vector_type(4)))  float  v4f;

union V16U { v16bf v; v8bf h[2]; };

#define HW4K 4096
#define CFULL 512
#define C8 64

// ---------------------------------------------------------------------------
// Bilinear resize 80x80 -> 64x64 (fp32), per (batch*channel) plane.
// resize(conv1x1(x)) == conv1x1(resize(x)) since 1x1 conv is per-pixel linear.
// ---------------------------------------------------------------------------
__global__ __launch_bounds__(256) void resize_kernel(
    const float* __restrict__ x, float* __restrict__ y, int nch)
{
    int idx = blockIdx.x * 256 + threadIdx.x;
    int total = nch * 64 * 64;
    if (idx >= total) return;
    int xo = idx & 63;
    int yo = (idx >> 6) & 63;
    int ch = idx >> 12;

    float sx = fmaxf((xo + 0.5f) * 1.25f - 0.5f, 0.0f);
    float sy = fmaxf((yo + 0.5f) * 1.25f - 0.5f, 0.0f);
    int x0 = (int)sx; if (x0 > 79) x0 = 79;
    int y0 = (int)sy; if (y0 > 79) y0 = 79;
    int x1 = x0 + 1;  if (x1 > 79) x1 = 79;
    int y1 = y0 + 1;  if (y1 > 79) y1 = 79;
    float wx = sx - (float)x0;
    float wy = sy - (float)y0;

    const float* p = x + (size_t)ch * 6400;
    float r0 = p[y0 * 80 + x0] * (1.0f - wx) + p[y0 * 80 + x1] * wx;
    float r1 = p[y1 * 80 + x0] * (1.0f - wx) + p[y1 * 80 + x1] * wx;
    y[idx] = r0 * (1.0f - wy) + r1 * wy;
}

// ---------------------------------------------------------------------------
// Tiled fp32 GEMM for 1x1 conv: y[o,p] = bias[o] + sum_c w[o,c] * x[b,c,p]
// 64x64 output tile per block, 4x4 register micro-tile per thread.
// MODE 0: fp32, (b, o, p) layout      (final output conv)
// MODE 1: bf16, (b, o, p) layout      (Hc, c-major)
// MODE 2: bf16, (b, p, o) layout      (Ft / Gt, K-contiguous rows for WMMA)
// ---------------------------------------------------------------------------
template <int MODE>
__global__ __launch_bounds__(256) void conv1x1_k(
    const float* __restrict__ x, const float* __restrict__ w,
    const float* __restrict__ bias, void* __restrict__ y,
    int C, int P, int Ofull)
{
    __shared__ float wt[16][64];
    __shared__ float xt[16][64];

    const int batch = blockIdx.z;
    const int o0 = blockIdx.y * 64;
    const int p0 = blockIdx.x * 64;
    const int tid = threadIdx.x;
    const int tx = tid & 15;
    const int ty = tid >> 4;

    float acc[4][4];
#pragma unroll
    for (int a = 0; a < 4; ++a)
#pragma unroll
        for (int b2 = 0; b2 < 4; ++b2) acc[a][b2] = 0.0f;

    const float* xb = x + (size_t)batch * C * P;

    for (int kk = 0; kk < C; kk += 16) {
        int idx = tid;
#pragma unroll
        for (int l = 0; l < 4; ++l) {
            int c_l = idx & 15;
            int o_l = idx >> 4;
            wt[c_l][o_l] = w[(size_t)(o0 + o_l) * C + kk + c_l];
            idx += 256;
        }
        idx = tid;
#pragma unroll
        for (int l = 0; l < 4; ++l) {
            int p_l = idx & 63;
            int k_l = idx >> 6;
            xt[k_l][p_l] = xb[(size_t)(kk + k_l) * P + p0 + p_l];
            idx += 256;
        }
        __syncthreads();
#pragma unroll
        for (int k = 0; k < 16; ++k) {
            float wv[4], xv[4];
#pragma unroll
            for (int a = 0; a < 4; ++a) wv[a] = wt[k][ty + 16 * a];
#pragma unroll
            for (int b2 = 0; b2 < 4; ++b2) xv[b2] = xt[k][tx + 16 * b2];
#pragma unroll
            for (int a = 0; a < 4; ++a)
#pragma unroll
                for (int b2 = 0; b2 < 4; ++b2)
                    acc[a][b2] = fmaf(wv[a], xv[b2], acc[a][b2]);
        }
        __syncthreads();
    }

#pragma unroll
    for (int a = 0; a < 4; ++a) {
        int o = o0 + ty + 16 * a;
        float bv = bias[o];
#pragma unroll
        for (int b2 = 0; b2 < 4; ++b2) {
            int p = p0 + tx + 16 * b2;
            float v = acc[a][b2] + bv;
            if (MODE == 0) {
                ((float*)y)[((size_t)batch * Ofull + o) * P + p] = v;
            } else if (MODE == 1) {
                ((__bf16*)y)[((size_t)batch * Ofull + o) * P + p] = (__bf16)v;
            } else {
                ((__bf16*)y)[((size_t)batch * P + p) * (size_t)Ofull + o] = (__bf16)v;
            }
        }
    }
}

// ---------------------------------------------------------------------------
// Flash-attention with bf16 WMMA.
//   S[m,n] = sum_c Ft[m,c]*Gt[n,c]   (16x32 per wave per super-iter)
//   online softmax over n (4096), shared across 8 waves via LDS
//   O[m,c] += P[m,n] * Hc[c,n]       (each wave owns a 64-channel C slice)
// Workgroup: 8 waves (256 threads) -> one 16-row m-tile of one batch.
// D/C layout (ISA 7.12.2): VGPR r, lane l -> M = r + 8*(l/16), N = l%16.
// ---------------------------------------------------------------------------
__global__ __launch_bounds__(256) void attn_kernel(
    const __bf16* __restrict__ Ft,   // (B, 4096, 64)  row = m, K-contiguous
    const __bf16* __restrict__ Gt,   // (B, 4096, 64)  row = n, K-contiguous
    const __bf16* __restrict__ Hc,   // (B, 512, 4096) row = c, n-contiguous
    float* __restrict__ O)           // (B, 512, 4096) c-major fp32
{
    const int b   = blockIdx.y;
    const int m0  = blockIdx.x * 16;
    const int tid = threadIdx.x;
    const int wv  = tid >> 5;      // wave id 0..7
    const int lane = tid & 31;
    const int nl  = lane & 15;
    const int hi  = lane >> 4;

    __shared__ __bf16 pb[8 * 16 * 32];   // per-subtile P, row-major (m,k) bf16
    __shared__ float  redm[8][16];       // cross-wave row-max partials
    __shared__ float  reds[8][16];       // cross-wave row-sum partials

    // A operand (Ft rows m0..m0+15). A layout: lane (M=nl, half hi):
    //   V0-3 = K 8hi..8hi+7, V4-7 = K 16+8hi..16+8hi+7 -> 16B contiguous runs.
    v16bf af0, af1;
    {
        const __bf16* fr = Ft + ((size_t)b * HW4K + m0 + nl) * C8;
        V16U u0, u1;
        u0.h[0] = *(const v8bf*)(fr + 8 * hi);
        u0.h[1] = *(const v8bf*)(fr + 16 + 8 * hi);
        u1.h[0] = *(const v8bf*)(fr + 32 + 8 * hi);
        u1.h[1] = *(const v8bf*)(fr + 48 + 8 * hi);
        af0 = u0.v;
        af1 = u1.v;
    }

    float mrun[8], lrun[8];
    v8f acc[4];
#pragma unroll
    for (int r = 0; r < 8; ++r) { mrun[r] = -3.0e38f; lrun[r] = 0.0f; }
#pragma unroll
    for (int ct = 0; ct < 4; ++ct)
#pragma unroll
        for (int r = 0; r < 8; ++r) acc[ct][r] = 0.0f;

    v8f z;
#pragma unroll
    for (int r = 0; r < 8; ++r) z[r] = 0.0f;

#pragma unroll 1
    for (int s = 0; s < 16; ++s) {          // 16 super-iters x 256 columns
        const int n0 = s * 256 + wv * 32;   // this wave's 32 columns

        // ---- S = Ft x Gt^T for 32 columns (two 16-col groups, K=64=2x32)
        // B layout: lane (N=nl, half hi): K = 32*kc + 8*hi + [0..7], 32*kc + 16 + 8*hi + [0..7].
        v8f s0, s1;
        {
            const __bf16* g0 = Gt + ((size_t)b * HW4K + n0 + nl) * C8;
            const __bf16* g1 = Gt + ((size_t)b * HW4K + n0 + 16 + nl) * C8;
            V16U b00, b10, b01, b11;
            b00.h[0] = *(const v8bf*)(g0 + 8 * hi);
            b00.h[1] = *(const v8bf*)(g0 + 16 + 8 * hi);
            b10.h[0] = *(const v8bf*)(g0 + 32 + 8 * hi);
            b10.h[1] = *(const v8bf*)(g0 + 48 + 8 * hi);
            b01.h[0] = *(const v8bf*)(g1 + 8 * hi);
            b01.h[1] = *(const v8bf*)(g1 + 16 + 8 * hi);
            b11.h[0] = *(const v8bf*)(g1 + 32 + 8 * hi);
            b11.h[1] = *(const v8bf*)(g1 + 48 + 8 * hi);
            s0 = __builtin_amdgcn_wmma_f32_16x16x32_bf16(false, af0, false, b00.v, (short)0, z,  false, false);
            s0 = __builtin_amdgcn_wmma_f32_16x16x32_bf16(false, af1, false, b10.v, (short)0, s0, false, false);
            s1 = __builtin_amdgcn_wmma_f32_16x16x32_bf16(false, af0, false, b01.v, (short)0, z,  false, false);
            s1 = __builtin_amdgcn_wmma_f32_16x16x32_bf16(false, af1, false, b11.v, (short)0, s1, false, false);
        }

        // ---- per-half row-max partials (rows 8*hi + r live across 16 lanes)
        float mx[8];
#pragma unroll
        for (int r = 0; r < 8; ++r) {
            float v = fmaxf(s0[r], s1[r]);
            v = fmaxf(v, __shfl_xor(v, 1, 32));
            v = fmaxf(v, __shfl_xor(v, 2, 32));
            v = fmaxf(v, __shfl_xor(v, 4, 32));
            v = fmaxf(v, __shfl_xor(v, 8, 32));
            mx[r] = v;
        }
        if (nl == 0) {
#pragma unroll
            for (int r = 0; r < 8; ++r) redm[wv][8 * hi + r] = mx[r];
        }
        __syncthreads();

        // ---- combine max across waves; rescale factor vs running max
        float alpha[8];
#pragma unroll
        for (int r = 0; r < 8; ++r) {
            float m2 = mrun[r];
#pragma unroll
            for (int w2 = 0; w2 < 8; ++w2) m2 = fmaxf(m2, redm[w2][8 * hi + r]);
            alpha[r] = __expf(mrun[r] - m2);
            mrun[r] = m2;
        }

        // ---- P = exp(S - max); write bf16 P tile; partial row sums
        float ps[8];
#pragma unroll
        for (int r = 0; r < 8; ++r) {
            float p0 = __expf(s0[r] - mrun[r]);
            float p1 = __expf(s1[r] - mrun[r]);
            int row = 8 * hi + r;
            pb[wv * 512 + row * 32 + nl]      = (__bf16)p0;
            pb[wv * 512 + row * 32 + 16 + nl] = (__bf16)p1;
            float t = p0 + p1;
            t += __shfl_xor(t, 1, 32);
            t += __shfl_xor(t, 2, 32);
            t += __shfl_xor(t, 4, 32);
            t += __shfl_xor(t, 8, 32);
            ps[r] = t;
        }
        if (nl == 0) {
#pragma unroll
            for (int r = 0; r < 8; ++r) reds[wv][8 * hi + r] = ps[r];
        }
        __syncthreads();

        // ---- update running denominator; rescale accumulators
#pragma unroll
        for (int r = 0; r < 8; ++r) {
            float ls = 0.0f;
#pragma unroll
            for (int w2 = 0; w2 < 8; ++w2) ls += reds[w2][8 * hi + r];
            lrun[r] = lrun[r] * alpha[r] + ls;
        }
#pragma unroll
        for (int ct = 0; ct < 4; ++ct)
#pragma unroll
            for (int r = 0; r < 8; ++r) acc[ct][r] *= alpha[r];

        // ---- O += P x H over all 8 subtiles; this wave's 64-channel slice
#pragma unroll
        for (int j = 0; j < 8; ++j) {
            V16U ap;   // A operand from LDS P tile (row nl, 16B runs)
            ap.h[0] = *(const v8bf*)&pb[j * 512 + nl * 32 + 8 * hi];
            ap.h[1] = *(const v8bf*)&pb[j * 512 + nl * 32 + 16 + 8 * hi];
            const int n0j = s * 256 + j * 32;
#pragma unroll
            for (int ct = 0; ct < 4; ++ct) {
                const __bf16* hr = Hc + ((size_t)b * CFULL + wv * 64 + ct * 16 + nl) * HW4K
                                      + n0j + 8 * hi;
                V16U bh;   // B operand: column c=nl, K = n in two 16B runs (8hi, 16+8hi)
                bh.h[0] = *(const v8bf*)(hr);
                bh.h[1] = *(const v8bf*)(hr + 16);
                acc[ct] = __builtin_amdgcn_wmma_f32_16x16x32_bf16(
                    false, ap.v, false, bh.v, (short)0, acc[ct], false, false);
            }
        }
    }

    // ---- epilogue: divide by softmax denominator, store O (c-major fp32)
    float inv[8];
#pragma unroll
    for (int r = 0; r < 8; ++r) inv[r] = 1.0f / lrun[r];
#pragma unroll
    for (int ct = 0; ct < 4; ++ct) {
        float* orow = O + ((size_t)b * CFULL + wv * 64 + ct * 16 + nl) * HW4K
                        + m0 + 8 * hi;
        v4f lo, hi4;
#pragma unroll
        for (int r = 0; r < 4; ++r) lo[r] = acc[ct][r] * inv[r];
#pragma unroll
        for (int r = 0; r < 4; ++r) hi4[r] = acc[ct][4 + r] * inv[4 + r];
        *(v4f*)orow = lo;
        *(v4f*)(orow + 4) = hi4;
    }
}

// ---------------------------------------------------------------------------
// Launch
// ---------------------------------------------------------------------------
extern "C" void kernel_launch(void* const* d_in, const int* in_sizes, int n_in,
                              void* d_out, int out_size, void* d_ws, size_t ws_size,
                              hipStream_t stream)
{
    const float* content = (const float*)d_in[0];   // (4, 512, 64, 64)
    const float* style   = (const float*)d_in[1];   // (4, 512, 80, 80)
    const float* f_w = (const float*)d_in[2];
    const float* f_b = (const float*)d_in[3];
    const float* g_w = (const float*)d_in[4];
    const float* g_b = (const float*)d_in[5];
    const float* h_w = (const float*)d_in[6];
    const float* h_b = (const float*)d_in[7];
    const float* out_w = (const float*)d_in[8];
    const float* out_b = (const float*)d_in[9];

    char* ws = (char*)d_ws;
    float*  style_r = (float*)(ws);                      // 33,554,432 B
    __bf16* Ft      = (__bf16*)(ws + 33554432);          //  2,097,152 B
    __bf16* Gt      = (__bf16*)(ws + 35651584);          //  2,097,152 B
    __bf16* Hc      = (__bf16*)(ws + 37748736);          // 16,777,216 B
    float*  Oat     = (float*)(ws + 54525952);           // 33,554,432 B (end 88,080,384)

    // 1. resize style 80x80 -> 64x64 (linear op commutes with 1x1 convs)
    resize_kernel<<<dim3((2048 * 64 * 64) / 256), 256, 0, stream>>>(style, style_r, 2048);

    // 2. conv1x1 projections -> bf16 operands for WMMA
    conv1x1_k<2><<<dim3(64, 1, 4), 256, 0, stream>>>(content, f_w, f_b, (void*)Ft, 512, 4096, 64);
    conv1x1_k<2><<<dim3(64, 1, 4), 256, 0, stream>>>(style_r, g_w, g_b, (void*)Gt, 512, 4096, 64);
    conv1x1_k<1><<<dim3(64, 8, 4), 256, 0, stream>>>(style_r, h_w, h_b, (void*)Hc, 512, 4096, 512);

    // 3. flash attention (WMMA bf16, fp32 accumulate)
    attn_kernel<<<dim3(256, 4), 256, 0, stream>>>(Ft, Gt, Hc, Oat);

    // 4. final 1x1 conv (fp32) -> d_out
    conv1x1_k<0><<<dim3(64, 8, 4), 256, 0, stream>>>(Oat, out_w, out_b, d_out, 512, 4096, 512);
}
